// MoETorch_37082747634691
// MI455X (gfx1250) — matrix operand order repaired and checked by reading, not compiled, not run on hardware
//
#include <hip/hip_runtime.h>

// ---------------- problem constants ----------------
#define D_H   2048   // hidden
#define D_E   1408   // per routed expert
#define D_S   2816   // shared expert dim
#define NE    8      // routed experts
#define T_TOK 2048   // tokens (B*S)
#define TM    16     // token tile per workgroup
#define FCHUNK 128   // f-dim chunk (8 waves * 16)

// LDS layout (dynamic): x tile bf16 pairs [16][1028] + h tile f32 [16][132] + tok/wt
#define XLDS_STRIDE 1028
#define HLDS_STRIDE 132
#define SMEM_BYTES (TM*XLDS_STRIDE*4 + TM*HLDS_STRIDE*4 + TM*4 + TM*4)

// workspace bf16-pair region layout (dword counts)
#define N_X    (T_TOK * (D_H/2))          //  2,097,152
#define N_WGE  (NE * D_E * (D_H/2))       // 11,534,336
#define N_WDE  (NE * D_H * (D_E/2))       // 11,534,336
#define N_WGS  (D_S * (D_H/2))            //  2,883,584
#define N_WDS  (D_H * (D_S/2))            //  2,883,584
#define WS_BF_OFF (1 << 18)               // bf16 region starts 256 KB into ws

typedef __attribute__((ext_vector_type(16))) __bf16 v16bf;
typedef __attribute__((ext_vector_type(8)))  float  v8f;

union BfPack { uint4 q[2]; unsigned int u[8]; v16bf v; };

static __device__ __forceinline__ unsigned int pack_bf16(float lo, float hi) {
  // round-to-nearest-even f32 -> bf16, packed pair
  unsigned a = __float_as_uint(lo);
  unsigned b = __float_as_uint(hi);
  a = (a + 0x7FFFu + ((a >> 16) & 1u)) >> 16;
  b = (b + 0x7FFFu + ((b >> 16) & 1u)) >> 16;
  return (a & 0xFFFFu) | (b << 16);
}

static __device__ __forceinline__ v8f zero8() {
  v8f z;
#pragma unroll
  for (int i = 0; i < 8; ++i) z[i] = 0.0f;
  return z;
}

// ---------------- kernel 0: f32 -> packed bf16 pairs (one pass over all weights + x) ----------------
__global__ void cvt_bf16_kernel(const float* __restrict__ src,
                                unsigned int* __restrict__ dst, long npairs) {
  long i = (long)blockIdx.x * blockDim.x + threadIdx.x;
  long stride = (long)gridDim.x * blockDim.x;
  long n2 = npairs >> 1;                      // two pairs (one float4) per step
  for (long p = i; p < n2; p += stride) {
    float4 f = *(const float4*)(src + 4 * p);
    dst[2 * p]     = pack_bf16(f.x, f.y);
    dst[2 * p + 1] = pack_bf16(f.z, f.w);
  }
}

// ---------------- kernel 1: router (softmax + top-2 dispatch lists) ----------------
__global__ void router_kernel(const float* __restrict__ x,
                              const float* __restrict__ gW,
                              int* __restrict__ cnt,
                              int* __restrict__ list,
                              float* __restrict__ wl) {
  int wave = threadIdx.x >> 5;
  int lane = threadIdx.x & 31;
  int tok  = blockIdx.x * 8 + wave;
  if (tok >= T_TOK) return;
  const float* xr = x + (size_t)tok * D_H;

  float p[NE];
#pragma unroll
  for (int e = 0; e < NE; ++e) p[e] = 0.0f;
  for (int i = lane; i < D_H; i += 32) {
    float xv = xr[i];
#pragma unroll
    for (int e = 0; e < NE; ++e) p[e] += xv * gW[e * D_H + i];
  }
#pragma unroll
  for (int e = 0; e < NE; ++e)
    for (int off = 16; off > 0; off >>= 1) p[e] += __shfl_down(p[e], off);

  if (lane == 0) {
    float mx = p[0];
#pragma unroll
    for (int e = 1; e < NE; ++e) mx = fmaxf(mx, p[e]);
    float sc[NE]; float s = 0.0f;
#pragma unroll
    for (int e = 0; e < NE; ++e) { sc[e] = __expf(p[e] - mx); s += sc[e]; }
    float inv = __builtin_amdgcn_rcpf(s);
    int b0 = 0; float v0 = -1.0f;
#pragma unroll
    for (int e = 0; e < NE; ++e) { if (sc[e] > v0) { v0 = sc[e]; b0 = e; } }
    int b1 = -1; float v1 = -1.0f;
#pragma unroll
    for (int e = 0; e < NE; ++e) { if (e != b0 && sc[e] > v1) { v1 = sc[e]; b1 = e; } }
    int s0 = atomicAdd(&cnt[b0], 1);
    list[b0 * T_TOK + s0] = tok;  wl[b0 * T_TOK + s0] = v0 * inv;
    int s1 = atomicAdd(&cnt[b1], 1);
    list[b1 * T_TOK + s1] = tok;  wl[b1 * T_TOK + s1] = v1 * inv;
  }
}

// ---------------- kernel 2: fused SiLU-gated expert FFN over a 16-token tile ----------------
// out[tok, :] += wt * ( silu(x@Wg^T) * (x@Wu^T) ) @ Wd^T   -- all operands pre-packed bf16 pairs
// Wg,Wu: [nf][D_H/2] dwords per expert; Wd: [D_H][nf/2] dwords per expert; x: [T][D_H/2] dwords.
__global__ void __launch_bounds__(256)
moe_ffn_tile(const unsigned int* __restrict__ x_bf,
             const unsigned int* __restrict__ Wg,
             const unsigned int* __restrict__ Wu,
             const unsigned int* __restrict__ Wd,
             float* __restrict__ out,
             const int* __restrict__ list,   // null => identity mapping (shared expert)
             const float* __restrict__ wl,
             const int* __restrict__ cnt,
             int nf, int tiles_per_expert, int use_list) {
  extern __shared__ unsigned char smem_raw[];
  unsigned int* x_lds = (unsigned int*)smem_raw;                       // [16][1028] bf16 pairs
  float* h_lds  = (float*)(smem_raw + TM * XLDS_STRIDE * 4);           // [16][132]
  int*   tok_lds = (int*)(smem_raw + TM * XLDS_STRIDE * 4 + TM * HLDS_STRIDE * 4);
  float* wt_lds  = (float*)(tok_lds + TM);

  const int expert = blockIdx.x / tiles_per_expert;
  const int tile   = blockIdx.x % tiles_per_expert;
  int nrow = TM;
  if (use_list) {
    int c = cnt[expert];
    if (tile * TM >= c) return;           // uniform per block: safe with barriers/WMMA
    nrow = c - tile * TM; if (nrow > TM) nrow = TM;
  }
  const unsigned int* Wg_e = Wg + (size_t)expert * nf * (D_H / 2);
  const unsigned int* Wu_e = Wu + (size_t)expert * nf * (D_H / 2);
  const unsigned int* Wd_e = Wd + (size_t)expert * D_H * (nf / 2);

  const int tid = threadIdx.x;
  if (tid < TM) {
    int tokv; float wv;
    if (use_list) {
      int base = expert * T_TOK + tile * TM;
      if (tid < nrow) { tokv = list[base + tid]; wv = wl[base + tid]; }
      else            { tokv = list[base];       wv = 0.0f; }   // pad: valid addr, zero weight
    } else {
      tokv = tile * TM + tid; wv = 1.0f;
    }
    tok_lds[tid] = tokv; wt_lds[tid] = wv;
  }
  __syncthreads();

  // Stage x tile (already bf16 pairs): straight b128 copy, no conversion VALU
  for (int idx = tid; idx < TM * (D_H / 8); idx += 256) {   // D_H/8 = 256 uint4 per row
    int r  = idx >> 8;
    int c4 = idx & 255;
    uint4 q = *(const uint4*)(x_bf + (size_t)tok_lds[r] * (D_H / 2) + 4 * c4);
    *(uint4*)(x_lds + r * XLDS_STRIDE + 4 * c4) = q;
  }
  __syncthreads();

  const int wave = tid >> 5;
  const int lane = tid & 31;
  const int m    = lane & 15;          // row (A) / col (B,C) index
  const int hi   = (lane >> 4) & 1;    // upper-lane half => +16 K offset (contiguous k order)

  // Down-projection accumulators: wave owns d in [256*wave, 256*wave+256) => 16 tiles of 16
  v8f acc[16];
#pragma unroll
  for (int t = 0; t < 16; ++t) acc[t] = zero8();

  for (int f0 = 0; f0 < nf; f0 += FCHUNK) {
    // ---- gate & up: C[16 tok x 16 f] over K = D_H, bf16 WMMA ----
    const int frow = f0 + wave * 16 + m;
    const unsigned int* wgrow = Wg_e + (size_t)frow * (D_H / 2);
    const unsigned int* wurow = Wu_e + (size_t)frow * (D_H / 2);
    v8f g = zero8(), u = zero8();
#pragma unroll 1
    for (int kk = 0; kk < D_H; kk += 32) {
      const int pb = (kk >> 1) + hi * 8;         // pair index base for this lane half
      BfPack A, Bg, Bu;
      {
        const unsigned int* xr = x_lds + m * XLDS_STRIDE;
        const int off = (kk >> 1) + hi * 8;
        A.q[0] = *(const uint4*)(xr + off);       // K = kk+{0..7} / {16..23}
        A.q[1] = *(const uint4*)(xr + off + 4);   // K = kk+{8..15} / {24..31}
      }
      Bg.q[0] = *(const uint4*)(wgrow + pb);
      Bg.q[1] = *(const uint4*)(wgrow + pb + 4);
      Bu.q[0] = *(const uint4*)(wurow + pb);
      Bu.q[1] = *(const uint4*)(wurow + pb + 4);
      g = __builtin_amdgcn_wmma_f32_16x16x32_bf16(false, A.v, false, Bg.v, (short)0, g, false, false);
      u = __builtin_amdgcn_wmma_f32_16x16x32_bf16(false, A.v, false, Bu.v, (short)0, u, false, false);
    }

    // ---- h = silu(g) * u -> stage chunk in LDS ----
    __syncthreads();   // previous chunk's down-phase reads of h_lds are done
#pragma unroll
    for (int i = 0; i < 8; ++i) {
      float gv = g[i];
      float hv = gv * __builtin_amdgcn_rcpf(1.0f + __expf(-gv)) * u[i];
      int row = i + 8 * hi;                       // C layout: M = vgpr(+8 for hi lanes)
      h_lds[row * HLDS_STRIDE + wave * 16 + m] = hv;
    }
    __syncthreads();

    // ---- down: acc[16 tok x 256 d] += H[16 x 128] @ Wd^T, K split in 4x32 ----
    const int dbase = wave * 256;
#pragma unroll 1
    for (int ks = 0; ks < 4; ++ks) {
      BfPack A;
      {
        const float* hr = h_lds + m * HLDS_STRIDE;
        const int kb = ks * 32 + (hi ? 16 : 0);
#pragma unroll
        for (int r = 0; r < 4; ++r)
          A.u[r] = pack_bf16(hr[kb + 2 * r], hr[kb + 2 * r + 1]);
#pragma unroll
        for (int r = 0; r < 4; ++r)
          A.u[4 + r] = pack_bf16(hr[kb + 8 + 2 * r], hr[kb + 8 + 2 * r + 1]);
      }
      const int fpb = (f0 >> 1) + ks * 16 + hi * 8;     // pair base within Wd row
#pragma unroll
      for (int t = 0; t < 16; ++t) {
        const unsigned int* wdrow = Wd_e + (size_t)(dbase + t * 16 + m) * (nf / 2);
        BfPack B;
        B.q[0] = *(const uint4*)(wdrow + fpb);
        B.q[1] = *(const uint4*)(wdrow + fpb + 4);
        acc[t] = __builtin_amdgcn_wmma_f32_16x16x32_bf16(false, A.v, false, B.v, (short)0, acc[t], false, false);
      }
    }
  }

  // ---- combine: out[tok, d] += wt[tok] * acc ----
#pragma unroll
  for (int t = 0; t < 16; ++t) {
    const int d = wave * 256 + t * 16 + m;
#pragma unroll
    for (int i = 0; i < 8; ++i) {
      int row = i + 8 * hi;
      float val = acc[t][i] * wt_lds[row];
      atomicAdd(out + (size_t)tok_lds[row] * D_H + d, val);
    }
  }
}

// ---------------- host launcher ----------------
extern "C" void kernel_launch(void* const* d_in, const int* in_sizes, int n_in,
                              void* d_out, int out_size, void* d_ws, size_t ws_size,
                              hipStream_t stream) {
  const float* x    = (const float*)d_in[0];
  const float* gW   = (const float*)d_in[1];
  const float* Wg_e = (const float*)d_in[2];
  const float* Wu_e = (const float*)d_in[3];
  const float* Wd_e = (const float*)d_in[4];
  const float* Wg_s = (const float*)d_in[5];
  const float* Wu_s = (const float*)d_in[6];
  const float* Wd_s = (const float*)d_in[7];
  float* out = (float*)d_out;

  // workspace: counters + dispatch lists (~131 KB), then bf16-pair tensors (~182 MB)
  unsigned char* wsb = (unsigned char*)d_ws;
  int*   cnt  = (int*)wsb;
  int*   list = cnt + NE;
  float* wl   = (float*)(list + NE * T_TOK);
  unsigned int* x_bf   = (unsigned int*)(wsb + WS_BF_OFF);
  unsigned int* Wg_bf  = x_bf   + N_X;
  unsigned int* Wu_bf  = Wg_bf  + N_WGE;
  unsigned int* Wd_bf  = Wu_bf  + N_WGE;
  unsigned int* Wgs_bf = Wd_bf  + N_WDE;
  unsigned int* Wus_bf = Wgs_bf + N_WGS;
  unsigned int* Wds_bf = Wus_bf + N_WGS;

  hipMemsetAsync(cnt, 0, NE * sizeof(int), stream);
  hipMemsetAsync(out, 0, (size_t)out_size * sizeof(float), stream);

  // one-pass f32 -> bf16-pair conversion (read 346 MB, write 173 MB; ~22 us at HBM speed)
  cvt_bf16_kernel<<<4096, 256, 0, stream>>>(x,    x_bf,   N_X);
  cvt_bf16_kernel<<<8192, 256, 0, stream>>>(Wg_e, Wg_bf,  N_WGE);
  cvt_bf16_kernel<<<8192, 256, 0, stream>>>(Wu_e, Wu_bf,  N_WGE);
  cvt_bf16_kernel<<<8192, 256, 0, stream>>>(Wd_e, Wd_bf,  N_WDE);
  cvt_bf16_kernel<<<4096, 256, 0, stream>>>(Wg_s, Wgs_bf, N_WGS);
  cvt_bf16_kernel<<<4096, 256, 0, stream>>>(Wu_s, Wus_bf, N_WGS);
  cvt_bf16_kernel<<<4096, 256, 0, stream>>>(Wd_s, Wds_bf, N_WDS);

  router_kernel<<<T_TOK / 8, 256, 0, stream>>>(x, gW, cnt, list, wl);

  // routed experts: grid covers worst case (all tokens on one expert); empty tiles exit early
  moe_ffn_tile<<<NE * (T_TOK / TM), 256, SMEM_BYTES, stream>>>(
      x_bf, Wg_bf, Wu_bf, Wd_bf, out, list, wl, cnt, D_E, T_TOK / TM, 1);

  // shared expert: identity token list, weight 1, nf = 2816
  moe_ffn_tile<<<T_TOK / TM, 256, SMEM_BYTES, stream>>>(
      x_bf, Wgs_bf, Wus_bf, Wds_bf, out, nullptr, nullptr, nullptr, D_S, T_TOK / TM, 0);
}
